// ResidualBlock_4329327034980
// MI455X (gfx1250) — matrix operand order repaired and checked by reading, not compiled, not run on hardware
//
#include <hip/hip_runtime.h>
#include <cstdint>

// ---------------------------------------------------------------------------
// CDNA5 (gfx1250) implementation of the CSDI-style ResidualBlock.
// All C>=128 contractions run on v_wmma_f32_16x16x32_bf16 (bf16 in, f32 acc),
// including the 550x550 cross-attention (WMMA flash attention).
// bf16 is the storage format for all GEMM operands; the GEMM inner loop is a
// pure b128 global->LDS copy feeding WMMA.
// ---------------------------------------------------------------------------

#define DEV static __device__ __forceinline__
typedef unsigned short bf16_t;

typedef __attribute__((ext_vector_type(16))) __bf16 v16bf;
typedef __attribute__((ext_vector_type(8)))  float  v8f;

struct Frag32 { uint4 lo, hi; };   // 32 bytes == one v16bf fragment

DEV bf16_t f2bf(float f) {                         // round-to-nearest-even
  unsigned int u = __float_as_uint(f);
  unsigned int r = 0x7FFFu + ((u >> 16) & 1u);
  return (bf16_t)((u + r) >> 16);
}
DEV float bf2f(bf16_t h) { return __uint_as_float((unsigned int)h << 16); }

DEV void unpack8(uint4 v, float* f) {
  union { uint4 v4; bf16_t us[8]; } u; u.v4 = v;
  #pragma unroll
  for (int j = 0; j < 8; ++j) f[j] = bf2f(u.us[j]);
}
DEV uint4 pack8(const float* f) {
  union { uint4 v4; bf16_t us[8]; } u;
  #pragma unroll
  for (int j = 0; j < 8; ++j) u.us[j] = f2bf(f[j]);
  return u.v4;
}

DEV float act_apply(float v, int act) {
  if (act == 1) return 0.5f * v * (1.0f + erff(v * 0.70710678118654752f)); // exact gelu
  if (act == 2) return v / (1.0f + __expf(-v));                            // silu
  return v;
}

// ---------------------------------------------------------------------------
// WMMA GEMM: out[M,N] = act(A[M,Kd] @ W[N,Kd]^T + bias[N]) + residual[M,N]
// A/W/residual/out are bf16; accumulation f32.
// 256 threads = 8 waves (2x4); block tile 64x128; wave tile 32x32 (4 accums).
// Guard-free: requires M%64==0, N%128==0, Kd%32==0.
// ---------------------------------------------------------------------------
#define BM 64
#define BN 128
#define BK 32
#define LDK 40   // bf16 row stride (80B, 16B-aligned, avoids bank conflicts)

__global__ __launch_bounds__(256) void k_gemm(
    const bf16_t* __restrict__ A, const bf16_t* __restrict__ Wt,
    const float* __restrict__ bias, const bf16_t* __restrict__ residual,
    bf16_t* __restrict__ out, int M, int N, int Kd, int act)
{
  __shared__ bf16_t As[BM][LDK];
  __shared__ bf16_t Bs[BN][LDK];
  const int tid   = threadIdx.x;
  const int wave  = tid >> 5;
  const int lane  = tid & 31;
  const int wm    = wave >> 2;        // 0..1 : 32-row group
  const int wn    = wave & 3;         // 0..3 : 32-col group
  const int tileM = blockIdx.y * BM;
  const int tileN = blockIdx.x * BN;
  const int sRow  = tid >> 2;         // 0..63
  const int sCol  = (tid & 3) << 3;   // 0,8,16,24 (bf16 elements)
  const int l15   = lane & 15;
  const int lh    = lane >> 4;

  v8f acc00 = {}, acc01 = {}, acc10 = {}, acc11 = {};

  const bf16_t* pa  = A  + (size_t)(tileM + sRow)      * Kd + sCol;
  const bf16_t* pb0 = Wt + (size_t)(tileN + sRow)      * Kd + sCol;
  const bf16_t* pb1 = Wt + (size_t)(tileN + 64 + sRow) * Kd + sCol;

  for (int k0 = 0; k0 < Kd; k0 += BK) {
    // pure b128 copies: zero VALU in the staging path
    *(uint4*)&As[sRow][sCol]      = *(const uint4*)(pa  + k0);
    *(uint4*)&Bs[sRow][sCol]      = *(const uint4*)(pb0 + k0);
    *(uint4*)&Bs[64 + sRow][sCol] = *(const uint4*)(pb1 + k0);
    if (k0 + BK < Kd) {               // gfx1250 global_prefetch_b8 on next K-tile
      __builtin_prefetch(pa  + k0 + BK, 0, 1);
      __builtin_prefetch(pb0 + k0 + BK, 0, 1);
      __builtin_prefetch(pb1 + k0 + BK, 0, 1);
    }
    __syncthreads();

    Frag32 t; const uint4* p;
    // A frags (16x32): lanes 0-15 K{0..7,16..23}, lanes 16-31 K{8..15,24..31}
    p = (const uint4*)&As[wm * 32 +      l15][lh * 8];  t.lo = p[0]; t.hi = p[2];
    v16bf a0 = __builtin_bit_cast(v16bf, t);
    p = (const uint4*)&As[wm * 32 + 16 + l15][lh * 8];  t.lo = p[0]; t.hi = p[2];
    v16bf a1 = __builtin_bit_cast(v16bf, t);
    // B frags (32x16): lanes 0-15 K{0..7,16..23}, lanes 16-31 K{8..15,24..31}
    p = (const uint4*)&Bs[wn * 32 +      l15][lh * 8]; t.lo = p[0]; t.hi = p[2];
    v16bf b0 = __builtin_bit_cast(v16bf, t);
    p = (const uint4*)&Bs[wn * 32 + 16 + l15][lh * 8]; t.lo = p[0]; t.hi = p[2];
    v16bf b1 = __builtin_bit_cast(v16bf, t);

    acc00 = __builtin_amdgcn_wmma_f32_16x16x32_bf16(false, a0, false, b0, (short)0, acc00, false, false);
    acc01 = __builtin_amdgcn_wmma_f32_16x16x32_bf16(false, a0, false, b1, (short)0, acc01, false, false);
    acc10 = __builtin_amdgcn_wmma_f32_16x16x32_bf16(false, a1, false, b0, (short)0, acc10, false, false);
    acc11 = __builtin_amdgcn_wmma_f32_16x16x32_bf16(false, a1, false, b1, (short)0, acc11, false, false);
    __syncthreads();
  }

  // C/D layout: vgpr j -> M = j + 8*(lane>=16), N = lane&15
  const int c0 = tileN + wn * 32 + l15;
  const int c1 = c0 + 16;
  const float bv0 = bias ? bias[c0] : 0.0f;
  const float bv1 = bias ? bias[c1] : 0.0f;
  #pragma unroll
  for (int j = 0; j < 8; ++j) {
    const int r0 = tileM + wm * 32 + lh * 8 + j;
    const int r1 = r0 + 16;
    float v00 = act_apply(acc00[j] + bv0, act);
    float v01 = act_apply(acc01[j] + bv1, act);
    float v10 = act_apply(acc10[j] + bv0, act);
    float v11 = act_apply(acc11[j] + bv1, act);
    if (residual) {
      v00 += bf2f(residual[(size_t)r0 * N + c0]);
      v01 += bf2f(residual[(size_t)r0 * N + c1]);
      v10 += bf2f(residual[(size_t)r1 * N + c0]);
      v11 += bf2f(residual[(size_t)r1 * N + c1]);
    }
    out[(size_t)r0 * N + c0] = f2bf(v00);
    out[(size_t)r0 * N + c1] = f2bf(v01);
    out[(size_t)r1 * N + c0] = f2bf(v10);
    out[(size_t)r1 * N + c1] = f2bf(v11);
  }
}

// f32 -> bf16 conversion (weights, once per launch)
__global__ void k_cvt(const float* __restrict__ in, bf16_t* __restrict__ out, int n)
{
  int idx = blockIdx.x * 256 + threadIdx.x;
  if (idx < n) out[idx] = f2bf(in[idx]);
}

// ---------------------------------------------------------------------------
// LayerNorm over C=128 (bf16 in -> bf16 out; one token per 128-thread block)
// ---------------------------------------------------------------------------
__global__ __launch_bounds__(128) void k_ln(const bf16_t* __restrict__ in,
    const float* __restrict__ g, const float* __restrict__ b,
    bf16_t* __restrict__ out)
{
  int row = blockIdx.x, c = threadIdx.x;
  __shared__ float red[128];
  float v = bf2f(in[(size_t)row * 128 + c]);
  red[c] = v; __syncthreads();
  for (int s = 64; s > 0; s >>= 1) { if (c < s) red[c] += red[c + s]; __syncthreads(); }
  float mean = red[0] * (1.0f / 128.0f);
  __syncthreads();
  float d = v - mean;
  red[c] = d * d; __syncthreads();
  for (int s = 64; s > 0; s >>= 1) { if (c < s) red[c] += red[c + s]; __syncthreads(); }
  float var = red[0] * (1.0f / 128.0f);
  out[(size_t)row * 128 + c] = f2bf(d * rsqrtf(var + 1e-5f) * g[c] + b[c]);
}

// diff[b,c] = emb[b,:] . Wd[c,:] + bd[c]   (f32 throughout, tiny)
__global__ __launch_bounds__(128) void k_diffproj(const float* __restrict__ emb,
    const float* __restrict__ Wd, const float* __restrict__ bd, float* __restrict__ diff)
{
  int b = blockIdx.x, c = threadIdx.x;
  float a = 0.f;
  for (int d = 0; d < 128; ++d) a += emb[b * 128 + d] * Wd[c * 128 + d];
  diff[b * 128 + c] = a + bd[c];
}

// y0[token,c] = LN_c( x[b,c,t] + diff[b,c] )   (token = b*550+t; f32 in, bf16 out)
__global__ __launch_bounds__(128) void k_add_diff_ln(const float* __restrict__ x,
    const float* __restrict__ diff, const float* __restrict__ g, const float* __restrict__ b,
    bf16_t* __restrict__ out)
{
  int i = blockIdx.x, c = threadIdx.x;
  int bb = i / 550, t = i - bb * 550;
  __shared__ float red[128];
  float v = x[((size_t)bb * 128 + c) * 550 + t] + diff[bb * 128 + c];
  red[c] = v; __syncthreads();
  for (int s = 64; s > 0; s >>= 1) { if (c < s) red[c] += red[c + s]; __syncthreads(); }
  float mean = red[0] * (1.0f / 128.0f);
  __syncthreads();
  float d = v - mean;
  red[c] = d * d; __syncthreads();
  for (int s = 64; s > 0; s >>= 1) { if (c < s) red[c] += red[c + s]; __syncthreads(); }
  float var = red[0] * (1.0f / 128.0f);
  out[(size_t)i * 128 + c] = f2bf(d * rsqrtf(var + 1e-5f) * g[c] + b[c]);
}

// ---------------------------------------------------------------------------
// Linformer compression: outc[seq,e,c] = sum_s in[seq*S+s,c]*EW[e,s] + Eb[e]
// ---------------------------------------------------------------------------
__global__ __launch_bounds__(256) void k_compress(const bf16_t* __restrict__ in,
    const float* __restrict__ EW, const float* __restrict__ Eb,
    bf16_t* __restrict__ outc, int S, int comp)
{
  int seq = blockIdx.x, tid = threadIdx.x;
  __shared__ float ew[1600];     // max comp*S = 32*50
  __shared__ float rowv[128];
  for (int i = tid; i < comp * S; i += 256) ew[i] = EW[i];
  float acc[16];
  #pragma unroll
  for (int j = 0; j < 16; ++j) acc[j] = 0.f;
  const int nent = comp * 128;
  __syncthreads();
  for (int s = 0; s < S; ++s) {
    if (tid < 128) rowv[tid] = bf2f(in[((size_t)seq * S + s) * 128 + tid]);
    __syncthreads();
    for (int j = 0; j < 16; ++j) {
      int idx = tid + (j << 8);
      if (idx < nent) acc[j] += rowv[idx & 127] * ew[(idx >> 7) * S + s];
    }
    __syncthreads();
  }
  for (int j = 0; j < 16; ++j) {
    int idx = tid + (j << 8);
    if (idx < nent)
      outc[((size_t)seq * comp + (idx >> 7)) * 128 + (idx & 127)] = f2bf(acc[j] + Eb[idx >> 7]);
  }
}

// ---------------------------------------------------------------------------
// Compressed attention per sequence: scores over e (<=32), per-head softmax,
// o[s,c] = sum_e P[h,e] * Vc[e,c]
// ---------------------------------------------------------------------------
__global__ __launch_bounds__(256) void k_lin_attn(const bf16_t* __restrict__ Q,
    const bf16_t* __restrict__ Kc, const bf16_t* __restrict__ Vc,
    bf16_t* __restrict__ O, int S, int comp, int causal)
{
  int seq = blockIdx.x, tid = threadIdx.x;
  __shared__ float kc[4096], vc[4096], qrow[128], sc[256], pmx[8], psum[8];
  const int n = comp * 128;
  for (int i = tid; i < n; i += 256) {
    kc[i] = bf2f(Kc[(size_t)seq * n + i]);
    vc[i] = bf2f(Vc[(size_t)seq * n + i]);
  }
  __syncthreads();
  for (int s = 0; s < S; ++s) {
    if (tid < 128) qrow[tid] = bf2f(Q[((size_t)seq * S + s) * 128 + tid]);
    __syncthreads();
    if (tid < 8 * comp) {
      int h = tid / comp, e = tid - h * comp;
      float a = 0.f;
      #pragma unroll
      for (int d = 0; d < 16; ++d) a += qrow[h * 16 + d] * kc[e * 128 + h * 16 + d];
      a *= 0.25f;                       // 1/sqrt(hd=16)
      if (causal && e < s) a = -1e10f;  // keep where e >= s (reference mask)
      sc[tid] = a;
    }
    __syncthreads();
    if (tid < 8) {
      float m = -3e38f;
      for (int e = 0; e < comp; ++e) m = fmaxf(m, sc[tid * comp + e]);
      float su = 0.f;
      for (int e = 0; e < comp; ++e) su += __expf(sc[tid * comp + e] - m);
      pmx[tid] = m; psum[tid] = su;
    }
    __syncthreads();
    if (tid < 8 * comp) {
      int h = tid / comp;
      sc[tid] = __expf(sc[tid] - pmx[h]) / psum[h];
    }
    __syncthreads();
    if (tid < 128) {
      int h = tid >> 4;
      float o = 0.f;
      for (int e = 0; e < comp; ++e) o += sc[h * comp + e] * vc[e * 128 + tid];
      O[((size_t)seq * S + s) * 128 + tid] = f2bf(o);
    }
    __syncthreads();
  }
}

// ---------------------------------------------------------------------------
// WMMA flash cross-attention. One block per (b, h, 128-query tile); each wave
// owns 16 query rows. 32 keys per iteration: 2 score WMMAs (K-dim = hd=16
// zero-padded to 32) + online softmax on fragments (4x shfl_xor row reduce)
// + 1 full-K P@V WMMA (P 16x32 via per-wave LDS tile, V staged transposed).
// ---------------------------------------------------------------------------
__global__ __launch_bounds__(256) void k_xattn(const bf16_t* __restrict__ Qa,
    const bf16_t* __restrict__ Ka, const bf16_t* __restrict__ Va,
    bf16_t* __restrict__ O, int T)
{
  const int b = blockIdx.z, h = blockIdx.y;
  const int tid = threadIdx.x, wave = tid >> 5, lane = tid & 31;
  const int l15 = lane & 15, lh = lane >> 4;
  const size_t base = (size_t)b * T * 128 + h * 16;

  __shared__ bf16_t ktile[32][16];     // K rows (t-local, d)
  __shared__ bf16_t vtile[16][32];     // V transposed (d, t-local)
  __shared__ bf16_t pbuf[8][16][40];   // per-wave P (s, t-local 0..31), padded
  __shared__ bf16_t zrow[16];          // 32B of zeros (padded K half)

  if (tid < 16) zrow[tid] = 0;

  // constant A fragment: Q rows, K-dim d zero-padded 16->32
  const int srow0 = blockIdx.x * 128 + wave * 16;
  int sq = srow0 + l15; if (sq > T - 1) sq = T - 1;
  Frag32 fq;
  fq.lo = *(const uint4*)(Qa + base + (size_t)sq * 128 + lh * 8);
  fq.hi = make_uint4(0u, 0u, 0u, 0u);
  v16bf aq = __builtin_bit_cast(v16bf, fq);

  v8f acc = {};
  float mrow[8], lrow[8];
  #pragma unroll
  for (int j = 0; j < 8; ++j) { mrow[j] = -3e38f; lrow[j] = 0.f; }

  const int nt = (T + 31) >> 5;
  for (int it = 0; it < nt; ++it) {
    const int t0 = it << 5;
    __syncthreads();
    if (tid < 64) {                      // stage 32 K rows
      int tt = tid >> 1, half = tid & 1, tg = t0 + tt;
      uint4 v = make_uint4(0u, 0u, 0u, 0u);
      if (tg < T) v = *(const uint4*)(Ka + base + (size_t)tg * 128 + half * 8);
      *(uint4*)&ktile[tt][half * 8] = v;
    } else if (tid < 128) {              // stage 32 V rows, transposed
      int q2 = tid - 64, tt = q2 >> 1, half = q2 & 1, tg = t0 + tt;
      uint4 v = make_uint4(0u, 0u, 0u, 0u);
      if (tg < T) v = *(const uint4*)(Va + base + (size_t)tg * 128 + half * 8);
      union { uint4 v4; bf16_t us[8]; } u; u.v4 = v;
      #pragma unroll
      for (int d = 0; d < 8; ++d) vtile[half * 8 + d][tt] = u.us[d];
    }
    __syncthreads();

    // two 16-col score tiles: S = Q @ K^T
    v8f z8 = {};
    Frag32 fk;
    const uint4* p0 = (const uint4*)&ktile[l15][lh * 8];
    fk.lo = p0[0]; fk.hi = make_uint4(0u, 0u, 0u, 0u);
    v8f sc0 = __builtin_amdgcn_wmma_f32_16x16x32_bf16(false, aq, false,
                __builtin_bit_cast(v16bf, fk), (short)0, z8, false, false);
    const uint4* p1 = (const uint4*)&ktile[16 + l15][lh * 8];
    fk.lo = p1[0]; fk.hi = make_uint4(0u, 0u, 0u, 0u);
    v8f sc1 = __builtin_amdgcn_wmma_f32_16x16x32_bf16(false, aq, false,
                __builtin_bit_cast(v16bf, fk), (short)0, z8, false, false);

    const bool ok0 = (t0 + l15 < T);
    const bool ok1 = (t0 + 16 + l15 < T);
    #pragma unroll
    for (int j = 0; j < 8; ++j) {          // row = j + 8*lh, col = l15 (+16)
      float s0 = ok0 ? sc0[j] * 0.25f : -1e10f;
      float s1 = ok1 ? sc1[j] * 0.25f : -1e10f;
      float rmax = fmaxf(s0, s1);
      rmax = fmaxf(rmax, __shfl_xor(rmax, 1, 32));
      rmax = fmaxf(rmax, __shfl_xor(rmax, 2, 32));
      rmax = fmaxf(rmax, __shfl_xor(rmax, 4, 32));
      rmax = fmaxf(rmax, __shfl_xor(rmax, 8, 32));
      float mn = fmaxf(mrow[j], rmax);
      float e0 = __expf(s0 - mn), e1 = __expf(s1 - mn);
      float rsum = e0 + e1;
      rsum += __shfl_xor(rsum, 1, 32);
      rsum += __shfl_xor(rsum, 2, 32);
      rsum += __shfl_xor(rsum, 4, 32);
      rsum += __shfl_xor(rsum, 8, 32);
      float corr = __expf(mrow[j] - mn);
      lrow[j] = lrow[j] * corr + rsum;
      mrow[j] = mn;
      acc[j] *= corr;
      pbuf[wave][(lh << 3) + j][l15]      = f2bf(e0);
      pbuf[wave][(lh << 3) + j][16 + l15] = f2bf(e1);
    }

    // o += P @ V  (full K=32: A = P 16x32, B = V^T 32x16)
    Frag32 fp;
    fp.lo = *(const uint4*)&pbuf[wave][l15][lh * 8];
    fp.hi = *(const uint4*)&pbuf[wave][l15][16 + lh * 8];
    Frag32 fv;
    fv.lo = *(const uint4*)&vtile[l15][lh * 8];
    fv.hi = *(const uint4*)&vtile[l15][16 + lh * 8];
    acc = __builtin_amdgcn_wmma_f32_16x16x32_bf16(false,
            __builtin_bit_cast(v16bf, fp), false,
            __builtin_bit_cast(v16bf, fv), (short)0, acc, false, false);
  }

  #pragma unroll
  for (int j = 0; j < 8; ++j) {
    int r = srow0 + (lh << 3) + j;
    if (r < T) O[base + (size_t)r * 128 + l15] = f2bf(acc[j] / lrow[j]);
  }
}

// ---------------------------------------------------------------------------
// Layout shuffles & elementwise epilogues (bf16, b128-vectorized over C)
// ---------------------------------------------------------------------------
// feat gather: dst token j=(b*50+l)*11+k  <-  src token i=b*550+k*50+l
__global__ void k_gather_feat(const bf16_t* __restrict__ y0, bf16_t* __restrict__ dst)
{
  int idx = blockIdx.x * 256 + threadIdx.x;     // 8-elem chunks
  if (idx >= 17600 * 16) return;
  int j = idx >> 4, c8 = (idx & 15) << 3;
  int b = j / 550, r = j - b * 550;
  int l = r / 11, k = r - l * 11;
  int i = b * 550 + k * 50 + l;
  *(uint4*)&dst[(size_t)j * 128 + c8] = *(const uint4*)&y0[(size_t)i * 128 + c8];
}

// y1 (time layout) = yt + gather(yf from feat layout)
__global__ void k_combine(const bf16_t* __restrict__ yt, const bf16_t* __restrict__ yf,
                          bf16_t* __restrict__ y1)
{
  int idx = blockIdx.x * 256 + threadIdx.x;     // 8-elem chunks
  if (idx >= 17600 * 16) return;
  int i = idx >> 4, c8 = (idx & 15) << 3;
  int b = i / 550, r = i - b * 550;
  int k = r / 50, l = r - k * 50;
  int j = (b * 50 + l) * 11 + k;
  float fa[8], fb[8];
  unpack8(*(const uint4*)&yt[(size_t)i * 128 + c8], fa);
  unpack8(*(const uint4*)&yf[(size_t)j * 128 + c8], fb);
  #pragma unroll
  for (int d = 0; d < 8; ++d) fa[d] += fb[d];
  *(uint4*)&y1[(size_t)i * 128 + c8] = pack8(fa);
}

// condbuf[(b*550+t), sd] = cond[b, sd, t]  (sd padded 144->160 with zeros)
__global__ void k_gather_cond(const float* __restrict__ cond, bf16_t* __restrict__ dst)
{
  int idx = blockIdx.x * 256 + threadIdx.x;
  if (idx >= 17600 * 160) return;
  int i = idx / 160, sd = idx - i * 160;
  int b = i / 550, t = i - b * 550;
  dst[idx] = (sd < 144) ? f2bf(cond[(size_t)b * 144 * 550 + (size_t)sd * 550 + t]) : (bf16_t)0;
}

// pad+convert Wc (128x144 f32) -> (128x160 bf16)
__global__ void k_pad_wc(const float* __restrict__ Wc, bf16_t* __restrict__ dst)
{
  int idx = blockIdx.x * 256 + threadIdx.x;
  if (idx >= 128 * 160) return;
  int n = idx / 160, k = idx - n * 160;
  dst[idx] = (k < 144) ? f2bf(Wc[n * 144 + k]) : (bf16_t)0;
}

// y3 = gamma*y2 + beta  (film is (M0,256): [:,0:128]=gamma, [:,128:256]=beta)
__global__ void k_film(const bf16_t* __restrict__ film, const bf16_t* __restrict__ y2,
                       bf16_t* __restrict__ y3)
{
  int idx = blockIdx.x * 256 + threadIdx.x;     // 8-elem chunks
  if (idx >= 17600 * 16) return;
  int i = idx >> 4, c8 = (idx & 15) << 3;
  float ga[8], be[8], yv[8];
  unpack8(*(const uint4*)&film[(size_t)i * 256 + c8], ga);
  unpack8(*(const uint4*)&film[(size_t)i * 256 + 128 + c8], be);
  unpack8(*(const uint4*)&y2[(size_t)i * 128 + c8], yv);
  #pragma unroll
  for (int d = 0; d < 8; ++d) yv[d] = ga[d] * yv[d] + be[d];
  *(uint4*)&y3[(size_t)i * 128 + c8] = pack8(yv);
}

// gate/filt split, act = sigmoid(gate)*tanh(filt); out = (x+act)/sqrt(2); d_out = [out | act]
__global__ void k_final(const bf16_t* __restrict__ z, const float* __restrict__ x,
                        float* __restrict__ out)
{
  int idx = blockIdx.x * 256 + threadIdx.x;
  if (idx >= 17600 * 128) return;
  int i = idx >> 7, c = idx & 127;
  int b = i / 550, t = i - b * 550;
  float gate = bf2f(z[(size_t)i * 256 + c]);
  float filt = bf2f(z[(size_t)i * 256 + 128 + c]);
  float a = (1.0f / (1.0f + __expf(-gate))) * tanhf(filt);
  size_t o = ((size_t)b * 128 + c) * 550 + t;
  out[o] = (x[o] + a) * 0.70710678118654752f;
  out[(size_t)2252800 + o] = a;
}

// ---------------------------------------------------------------------------
// Host orchestration
// ---------------------------------------------------------------------------
struct LinW { const bf16_t *Wq, *Wk, *Wv, *Wo, *W1, *W2; };
struct LinScratch { bf16_t *h, *Qb, *Kb, *Vb, *Kc, *Vc, *Ob, *x1, *x2, *ff, *out; };

static void run_linformer(void* const* d_in, int pbase, const LinW& w, const bf16_t* xin,
                          int Nseq, int S, int comp, int causal,
                          const LinScratch& s, hipStream_t stream)
{
  auto in = [&](int i) { return (const float*)d_in[i]; };
  const int M = Nseq * S;   // always 17600 here (divisible by BM)
  auto gemm = [&](const bf16_t* A, const bf16_t* W, const float* bias, const bf16_t* res,
                  bf16_t* out, int N, int Kd, int act) {
    dim3 g(N / BN, M / BM);
    k_gemm<<<g, 256, 0, stream>>>(A, W, bias, res, out, M, N, Kd, act);
  };
  k_ln<<<M, 128, 0, stream>>>(xin, in(pbase + 0), in(pbase + 1), s.h);
  gemm(s.h, w.Wq, nullptr, nullptr, s.Qb, 128, 128, 0);
  gemm(s.h, w.Wk, nullptr, nullptr, s.Kb, 128, 128, 0);
  gemm(s.h, w.Wv, nullptr, nullptr, s.Vb, 128, 128, 0);
  k_compress<<<Nseq, 256, 0, stream>>>(s.Kb, in(pbase + 5), in(pbase + 6), s.Kc, S, comp); // E
  k_compress<<<Nseq, 256, 0, stream>>>(s.Vb, in(pbase + 7), in(pbase + 8), s.Vc, S, comp); // F
  k_lin_attn<<<Nseq, 256, 0, stream>>>(s.Qb, s.Kc, s.Vc, s.Ob, S, comp, causal);
  gemm(s.Ob, w.Wo, in(pbase + 10), s.h, s.x1, 128, 128, 0);             // Wo + residual h
  k_ln<<<M, 128, 0, stream>>>(s.x1, in(pbase + 11), in(pbase + 12), s.x2);
  gemm(s.x2, w.W1, in(pbase + 14), nullptr, s.ff, 256, 128, 1);         // W1 + gelu
  gemm(s.ff, w.W2, in(pbase + 16), s.x2, s.out, 128, 256, 0);           // W2 + residual x2
}

// input-index map (setup_inputs dict order, nested dicts depth-first)
enum {
  IN_X = 0, IN_COND = 1, IN_DEMB = 2, IN_WD = 3, IN_BD = 4,
  T_BASE = 5,   // time linformer: 17 entries
  F_BASE = 22,  // feat linformer: 17 entries
  IN_N1G = 39, IN_N1B = 40, IN_N2G = 41, IN_N2B = 42,
  IN_WC = 43, IN_BC = 44,
  IN_WAQ = 45, IN_BAQ = 46, IN_WAK = 47, IN_BAK = 48, IN_WAV = 49, IN_BAV = 50,
  IN_WAO = 51, IN_BAO = 52,
  IN_WF1 = 53, IN_BF1 = 54, IN_WF2 = 55, IN_BF2 = 56,
  IN_WMID = 57, IN_BMID = 58, IN_WOUT = 59, IN_BOUT = 60
};

extern "C" void kernel_launch(void* const* d_in, const int* in_sizes, int n_in,
                              void* d_out, int out_size, void* d_ws, size_t ws_size,
                              hipStream_t stream)
{
  (void)in_sizes; (void)n_in; (void)out_size; (void)ws_size;
  const int Bb = 32, Cc = 128, SIDEP = 160;
  const int M0 = 32 * 550;
  auto in = [&](int i) { return (const float*)d_in[i]; };

  // byte arena in workspace (bf16 activations; peak ~70 MB)
  char* base = (char*)d_ws;
  size_t off = 0;
  auto allocB = [&](size_t bytes) { char* p = base + off; off = (off + bytes + 255) & ~(size_t)255; return p; };
  auto allocU = [&](size_t n) { return (bf16_t*)allocB(n * sizeof(bf16_t)); };
  auto allocF = [&](size_t n) { return (float*)allocB(n * sizeof(float)); };

  const size_t TOK = (size_t)M0 * Cc;
  bf16_t* S0 = allocU(TOK); bf16_t* S1 = allocU(TOK); bf16_t* S2 = allocU(TOK);
  bf16_t* S3 = allocU(TOK); bf16_t* S4 = allocU(TOK); bf16_t* S5 = allocU(TOK);
  bf16_t* S6 = allocU(TOK);
  bf16_t* T0 = allocU(2 * TOK); bf16_t* T1 = allocU(2 * TOK);
  bf16_t* U0 = allocU(TOK);     bf16_t* U1 = allocU(TOK);     // compressed K/V
  bf16_t* V0 = allocU((size_t)M0 * SIDEP);                    // padded cond activations
  float*  diffb = allocF((size_t)Bb * Cc);

  // one-time per-launch weight conversion f32 -> bf16
  auto cvtW = [&](int idx, size_t n) -> const bf16_t* {
    bf16_t* p = allocU(n);
    k_cvt<<<((int)n + 255) / 256, 256, 0, stream>>>(in(idx), p, (int)n);
    return p;
  };
  auto cvtLin = [&](int pb) {
    LinW w;
    w.Wq = cvtW(pb + 2, 16384);  w.Wk = cvtW(pb + 3, 16384);
    w.Wv = cvtW(pb + 4, 16384);  w.Wo = cvtW(pb + 9, 16384);
    w.W1 = cvtW(pb + 13, 32768); w.W2 = cvtW(pb + 15, 32768);
    return w;
  };
  LinW wT = cvtLin(T_BASE), wF = cvtLin(F_BASE);
  const bf16_t* Wmid = cvtW(IN_WMID, 16384);
  const bf16_t* Waq  = cvtW(IN_WAQ, 16384);
  const bf16_t* Wak  = cvtW(IN_WAK, 16384);
  const bf16_t* Wav  = cvtW(IN_WAV, 16384);
  const bf16_t* Wao  = cvtW(IN_WAO, 16384);
  const bf16_t* Wf1  = cvtW(IN_WF1, 32768);
  const bf16_t* Wf2  = cvtW(IN_WF2, 65536);
  const bf16_t* Wout = cvtW(IN_WOUT, 32768);
  bf16_t* WcP = allocU((size_t)128 * SIDEP);
  k_pad_wc<<<(128 * SIDEP + 255) / 256, 256, 0, stream>>>(in(IN_WC), WcP);

  auto gemm = [&](const bf16_t* A, const bf16_t* W, const float* bias, const bf16_t* res,
                  bf16_t* out, int N, int Kd, int act) {
    dim3 g(N / BN, M0 / BM);
    k_gemm<<<g, 256, 0, stream>>>(A, W, bias, res, out, M0, N, Kd, act);
  };
  const int CH_GRID = (M0 * 16 + 255) / 256;     // 8-elem chunk grids
  const int EW_GRID = (M0 * 128 + 255) / 256;

  // 1) diffusion projection + add + LN(n1) -> y0 (token-major bf16) in S0
  k_diffproj<<<Bb, 128, 0, stream>>>(in(IN_DEMB), in(IN_WD), in(IN_BD), diffb);
  k_add_diff_ln<<<M0, 128, 0, stream>>>(in(IN_X), diffb, in(IN_N1G), in(IN_N1B), S0);

  // 2) time linformer (Nseq=352, S=50, comp=32, causal) -> outT in S1
  { LinScratch st{S1, S2, S3, S4, U0, U1, S3, S2, S4, T0, S1};
    run_linformer(d_in, T_BASE, wT, S0, 352, 50, 32, 1, st, stream); }

  // 3) feature linformer (Nseq=1600, S=11, comp=11) -> outF in S0
  k_gather_feat<<<CH_GRID, 256, 0, stream>>>(S0, S2);
  { LinScratch sf{S0, S3, S5, S6, U0, U1, S5, S3, S6, T0, S0};
    run_linformer(d_in, F_BASE, wF, S2, 1600, 11, 11, 0, sf, stream); }

  // 4) combine, mid projection, LN(n2)
  k_combine<<<CH_GRID, 256, 0, stream>>>(S1, S0, S2);                 // y1 in S2
  gemm(S2, Wmid, in(IN_BMID), nullptr, S3, 128, 128, 0);             // ymid in S3
  k_ln<<<M0, 128, 0, stream>>>(S3, in(IN_N2G), in(IN_N2B), S4);      // y2 in S4

  // 5) cond projection (Kd padded to 160), cross-attention Q/K/V
  k_gather_cond<<<(M0 * SIDEP + 255) / 256, 256, 0, stream>>>(in(IN_COND), V0);
  gemm(V0, WcP, in(IN_BC),  nullptr, S5, 128, 160, 0);               // cproj in S5
  gemm(S4, Waq, in(IN_BAQ), nullptr, S1, 128, 128, 0);               // Qa in S1
  gemm(S5, Wak, in(IN_BAK), nullptr, S6, 128, 128, 0);               // Ka in S6
  gemm(S5, Wav, in(IN_BAV), nullptr, S0, 128, 128, 0);               // Va in S0

  // 6) WMMA flash cross-attention over T=550 -> ao in S3
  { dim3 g((550 + 127) / 128, 8, 32);
    k_xattn<<<g, 256, 0, stream>>>(S1, S6, S0, S3, 550); }

  // 7) output projection + FiLM
  gemm(S3, Wao, in(IN_BAO), nullptr, S2, 128, 128, 0);               // ao2 in S2
  gemm(S2, Wf1, in(IN_BF1), nullptr, T0, 256, 128, 2);               // silu -> f1 in T0
  gemm(T0, Wf2, in(IN_BF2), nullptr, T1, 256, 256, 0);               // film in T1
  k_film<<<CH_GRID, 256, 0, stream>>>(T1, S4, S5);                   // y3 in S5

  // 8) gate/filter projection + final activation / residual
  gemm(S5, Wout, in(IN_BOUT), nullptr, T0, 256, 128, 0);             // z in T0
  k_final<<<EW_GRID, 256, 0, stream>>>(T0, in(IN_X), (float*)d_out);
}
